// VMA_56349970923772
// MI455X (gfx1250) — matrix operand order repaired and checked by reading, not compiled, not run on hardware
//
#include <hip/hip_runtime.h>
#include <hip/hip_bf16.h>
#include <stdint.h>

typedef __attribute__((ext_vector_type(16))) __bf16 v16bf;
typedef __attribute__((ext_vector_type(8)))  float  v8f;

#define BG 16
#define CGN 64
#define DDIM 16
#define HDIM 64
#define WDIM 64
#define HW 4096        // HDIM*WDIM
#define DHW 65536      // DDIM*HW
#define SPOOL 144      // D + H + W
#define ROWPITCH 66    // w = -1 .. 64
#define SMEM_ROWS 12   // 3 kd x 4 h rows
#define K4_SMEM_BYTES (SMEM_ROWS * ROWPITCH * 64 * 2)   // 101376

__device__ __forceinline__ unsigned short f2bf(float f) {
    union { float f; uint32_t u; } x; x.f = f;
    uint32_t u = x.u;
    uint32_t r = u + 0x7FFFu + ((u >> 16) & 1u);   // round-to-nearest-even
    if ((u & 0x7F800000u) == 0x7F800000u) r = u;   // preserve Inf/NaN
    return (unsigned short)(r >> 16);
}

// ---------------------------------------------------------------------------
// K0: repack w3 (O=64, I=64, 3,3,3) f32 -> WMMA A-fragment bf16 layout.
// Fragment id = (tap*2 + kk)*4 + otile; each fragment = 32 lanes x 16 bf16.
// 16-bit A 16x32 layout (contiguous, matches B): lane<16 row M=lane,
// dword j -> K=2j,2j+1 ; lane>=16 adds +16 to K.
// ---------------------------------------------------------------------------
__global__ void k0_prep_afrag(const float* __restrict__ w3,
                              unsigned short* __restrict__ afrag) {
    int fid = blockIdx.x;          // 0..215
    int l   = threadIdx.x;         // 0..31
    int mo  = fid & 3;
    int kk  = (fid >> 2) & 1;
    int tap = fid >> 3;            // 0..26
    int o   = mo * 16 + (l & 15);
    unsigned short* dst = afrag + (size_t)fid * 512 + l * 16;
#pragma unroll
    for (int j = 0; j < 8; ++j) {
        int kbase = ((l >> 4) << 4) + (j << 1);
        int i0 = kk * 32 + kbase;
        dst[j * 2 + 0] = f2bf(w3[(size_t)(o * 64 + i0)     * 27 + tap]);
        dst[j * 2 + 1] = f2bf(w3[(size_t)(o * 64 + i0 + 1) * 27 + tap]);
    }
}

// ---------------------------------------------------------------------------
// K0b: one-time f32 -> bf16 conversion of x (so the conv streams bf16).
// ---------------------------------------------------------------------------
__global__ void __launch_bounds__(256)
k0b_tobf(const float* __restrict__ x, unsigned short* __restrict__ xbf) {
    size_t i = ((size_t)blockIdx.x * 256 + threadIdx.x) * 8;
    const float4* p = (const float4*)(x + i);
    float4 a = p[0], b = p[1];
    union { unsigned short s[8]; uint4 q; } o;
    o.s[0] = f2bf(a.x); o.s[1] = f2bf(a.y); o.s[2] = f2bf(a.z); o.s[3] = f2bf(a.w);
    o.s[4] = f2bf(b.x); o.s[5] = f2bf(b.y); o.s[6] = f2bf(b.z); o.s[7] = f2bf(b.w);
    *(uint4*)(xbf + i) = o.q;
}

// ---------------------------------------------------------------------------
// K1: directional pools per (bg,cg) slice -> pools[bc][144] (d:16, h:64, w:64)
// ---------------------------------------------------------------------------
__global__ void __launch_bounds__(256)
k1_pools(const float* __restrict__ x, float* __restrict__ pools) {
    int bc = blockIdx.x;                       // bg*64+cg
    const float* src = x + (size_t)bc * DHW;
    int t = threadIdx.x;
    __shared__ float accd[16], acch[64], accw[64];
    if (t < 16) accd[t] = 0.f;
    if (t < 64) { acch[t] = 0.f; accw[t] = 0.f; }
    __syncthreads();
    float wacc[16];
#pragma unroll
    for (int j = 0; j < 16; ++j) wacc[j] = 0.f;
    float hacc = 0.f;
#pragma unroll
    for (int k = 0; k < 16; ++k) {
        const float4* p = (const float4*)(src + (size_t)(t + 256 * k) * 16);
        float dsum = 0.f;
#pragma unroll
        for (int q = 0; q < 4; ++q) {
            float4 v = p[q];
            wacc[q * 4 + 0] += v.x; wacc[q * 4 + 1] += v.y;
            wacc[q * 4 + 2] += v.z; wacc[q * 4 + 3] += v.w;
            dsum += v.x + v.y + v.z + v.w;
        }
        hacc += dsum;
        atomicAdd(&accd[k], dsum);
    }
    atomicAdd(&acch[t >> 2], hacc);
    int wq = (t & 3) * 16;
#pragma unroll
    for (int j = 0; j < 16; ++j) atomicAdd(&accw[wq + j], wacc[j]);
    __syncthreads();
    float* dst = pools + (size_t)bc * SPOOL;
    if (t < 16)        dst[t] = accd[t]      * (1.0f / 4096.f);
    else if (t < 80)   dst[t] = acch[t - 16] * (1.0f / 1024.f);
    else if (t < 144)  dst[t] = accw[t - 80] * (1.0f / 1024.f);
}

// ---------------------------------------------------------------------------
// K2: hw = w1 @ cat + b1, sigmoid -> sig[bc][144]
// ---------------------------------------------------------------------------
__global__ void __launch_bounds__(256)
k2_mix(const float* __restrict__ pools, const float* __restrict__ w1,
       const float* __restrict__ b1, float* __restrict__ sig) {
    int bg = blockIdx.x;
    int t = threadIdx.x;
    __shared__ float catS[64][SPOOL];
    __shared__ float w1S[64][64];
    for (int i = t; i < 64 * SPOOL; i += 256)
        catS[i / SPOOL][i % SPOOL] = pools[(size_t)bg * 64 * SPOOL + i];
    for (int i = t; i < 64 * 64; i += 256) w1S[i >> 6][i & 63] = w1[i];
    __syncthreads();
    for (int m = t; m < 64 * SPOOL; m += 256) {
        int o = m / SPOOL, s = m % SPOOL;
        float acc = b1[o];
#pragma unroll 8
        for (int c = 0; c < 64; ++c) acc += w1S[o][c] * catS[c][s];
        sig[(size_t)bg * 64 * SPOOL + m] = 1.0f / (1.0f + __expf(-acc));
    }
}

// ---------------------------------------------------------------------------
// K3: per-channel mean / rstd of gated tensor (gate rebuilt from sig tables)
// ---------------------------------------------------------------------------
__global__ void __launch_bounds__(256)
k3_stats(const float* __restrict__ x, const float* __restrict__ sig,
         float* __restrict__ mu, float* __restrict__ rstd) {
    int bc = blockIdx.x;
    int t = threadIdx.x;
    const float* src = x + (size_t)bc * DHW;
    const float* sg = sig + (size_t)bc * SPOOL;
    __shared__ float sgd[16], sgh[64], rsum[2];
    if (t < 16) sgd[t] = sg[t];
    if (t < 64) sgh[t] = sg[16 + t];
    if (t < 2)  rsum[t] = 0.f;
    __syncthreads();
    float sw_ = sg[80 + (t & 63)];
    float s1 = 0.f, s2 = 0.f;
    for (int k = 0; k < 256; ++k) {
        int idx = t + 256 * k;
        int d = idx >> 12;
        int h = (idx >> 6) & 63;
        float g = src[idx] * sgd[d] * sgh[h] * sw_;
        s1 += g; s2 += g * g;
    }
    atomicAdd(&rsum[0], s1);
    atomicAdd(&rsum[1], s2);
    __syncthreads();
    if (t == 0) {
        float m = rsum[0] * (1.0f / DHW);
        float v = rsum[1] * (1.0f / DHW) - m * m;
        mu[bc] = m;
        rstd[bc] = rsqrtf(v + 1e-5f);
    }
}

// ---------------------------------------------------------------------------
// K4: 3x3x3 conv as 27 shifted 64x64 channel GEMMs, bf16 WMMA.
// One WG per (bg, d0); 32 iterations, each staging 12 halo rows (2 output
// h-rows) into 101 KB dynamic LDS, then two low-pressure compute passes
// (hrow = 0,1).  Wave = (otile, w-half): 2 accumulators, 2 A frags and 4 B
// frags live per tap -> in-place v_wmma accumulation, no spill shuffling.
// 108 WMMA per wave per hrow pass. x2 (f32) written into d_out scratch.
// ---------------------------------------------------------------------------
extern __shared__ __align__(16) unsigned short smem4[];

__global__ void __launch_bounds__(256)
k4_conv(const unsigned short* __restrict__ xbf,
        const unsigned short* __restrict__ afrag,
        const float* __restrict__ b3, float* __restrict__ x2) {
    int bg = blockIdx.x >> 4;
    int d0 = blockIdx.x & 15;
    int t = threadIdx.x;
    int wv = t >> 5;
    int l = t & 31;
    int otile = wv & 3;
    int ngrp = wv >> 2;                    // 0/1 -> w cols [0,32) / [32,64)
    int nlo = l & 15;
    int chsel = (l < 16) ? 0 : 16;         // contiguous K packing (K = 16h + s, matches A frags)
    const unsigned short* xb = xbf + (size_t)bg * 64 * DHW;
    union AF { uint4 q[2]; v16bf v; };
    union CF { v8f v; float f[8]; };

    for (int it = 0; it < 32; ++it) {
        int h0 = it * 2;
        __syncthreads();
        // ---- stage 12 rows x 64 ch x 66 w (zero halo at w=-1,64) ----
        for (int task = t; task < SMEM_ROWS * 32; task += 256) {
            int r = task >> 5;
            int chp = (task & 31) * 2;     // channel pair
            int din = d0 + (r >> 2) - 1;
            int hin = h0 + (r & 3) - 1;
            uint32_t* dsrow = (uint32_t*)&smem4[(size_t)r * ROWPITCH * 64] + (chp >> 1);
            if (din >= 0 && din < DDIM && hin >= 0 && hin < HDIM) {
                const uint4* ga = (const uint4*)(xb + (size_t)chp * DHW + din * HW + hin * WDIM);
                const uint4* gb = (const uint4*)(xb + (size_t)(chp + 1) * DHW + din * HW + hin * WDIM);
                dsrow[0] = 0u;
                dsrow[65 * 32] = 0u;
#pragma unroll
                for (int seg = 0; seg < 8; ++seg) {
                    uint4 a = ga[seg], b = gb[seg];
                    const unsigned short* pa = (const unsigned short*)&a;
                    const unsigned short* pb = (const unsigned short*)&b;
#pragma unroll
                    for (int j = 0; j < 8; ++j)
                        dsrow[(seg * 8 + j + 1) * 32] =
                            (uint32_t)pa[j] | ((uint32_t)pb[j] << 16);
                }
            } else {
#pragma unroll
                for (int q = 0; q < 66; ++q) dsrow[q * 32] = 0u;
            }
        }
        __syncthreads();
        // ---- hint-prefetch next iteration's staging rows ----
        if (it + 1 < 32) {
            int r = t >> 5;
            int chp = (t & 31) * 2;
            int din = d0 + (r >> 2) - 1;
            int hin = (h0 + 2) + (r & 3) - 1;
            if (r < SMEM_ROWS && din >= 0 && din < DDIM && hin >= 0 && hin < HDIM)
                __builtin_prefetch(xb + (size_t)chp * DHW + din * HW + hin * WDIM, 0, 1);
        }
        // ---- compute: two output h-rows from the staged window ----
        for (int hrow = 0; hrow < 2; ++hrow) {
            CF c0, c1;
#pragma unroll
            for (int rr = 0; rr < 8; ++rr) { c0.f[rr] = 0.f; c1.f[rr] = 0.f; }

            for (int tap = 0; tap < 27; ++tap) {
                int kw = tap % 3;
                int khd = tap / 3;         // kd*3 + kh
                int r = (khd / 3) * 4 + (khd % 3) + hrow;
                int wpos0 = ngrp * 32 + nlo + kw;
                int wpos1 = wpos0 + 16;
                AF a0, a1, b0, b1;
                const uint4* ap0 = (const uint4*)(afrag + ((size_t)((tap * 2 + 0) * 4 + otile)) * 512 + l * 16);
                const uint4* ap1 = (const uint4*)(afrag + ((size_t)((tap * 2 + 1) * 4 + otile)) * 512 + l * 16);
                // kk = 0 half (channels 0..31)
                a0.q[0] = ap0[0]; a0.q[1] = ap0[1];
                {
                    const uint4* bp = (const uint4*)&smem4[((size_t)r * ROWPITCH + wpos0) * 64 + chsel];
                    b0.q[0] = bp[0]; b0.q[1] = bp[1];
                }
                {
                    const uint4* bp = (const uint4*)&smem4[((size_t)r * ROWPITCH + wpos1) * 64 + chsel];
                    b1.q[0] = bp[0]; b1.q[1] = bp[1];
                }
                c0.v = __builtin_amdgcn_wmma_f32_16x16x32_bf16(
                    false, a0.v, false, b0.v, (short)0, c0.v, false, false);
                c1.v = __builtin_amdgcn_wmma_f32_16x16x32_bf16(
                    false, a0.v, false, b1.v, (short)0, c1.v, false, false);
                // kk = 1 half (channels 32..63)
                a1.q[0] = ap1[0]; a1.q[1] = ap1[1];
                {
                    const uint4* bp = (const uint4*)&smem4[((size_t)r * ROWPITCH + wpos0) * 64 + 32 + chsel];
                    b0.q[0] = bp[0]; b0.q[1] = bp[1];
                }
                {
                    const uint4* bp = (const uint4*)&smem4[((size_t)r * ROWPITCH + wpos1) * 64 + 32 + chsel];
                    b1.q[0] = bp[0]; b1.q[1] = bp[1];
                }
                c0.v = __builtin_amdgcn_wmma_f32_16x16x32_bf16(
                    false, a1.v, false, b0.v, (short)0, c0.v, false, false);
                c1.v = __builtin_amdgcn_wmma_f32_16x16x32_bf16(
                    false, a1.v, false, b1.v, (short)0, c1.v, false, false);
            }
            // ---- store x2 row (+bias). C layout: VGPR rr -> M=rr / rr+8 ----
            int h = h0 + hrow;
            int ncol0 = ngrp * 32 + nlo;
            int mbase = otile * 16 + ((l >> 4) << 3);
            float* ob = x2 + (size_t)bg * 64 * DHW + (size_t)d0 * HW + (size_t)h * WDIM;
#pragma unroll
            for (int rr = 0; rr < 8; ++rr) {
                int o = mbase + rr;
                float bias = b3[o];
                ob[(size_t)o * DHW + ncol0]      = c0.f[rr] + bias;
                ob[(size_t)o * DHW + ncol0 + 16] = c1.f[rr] + bias;
            }
        }
    }
}

// ---------------------------------------------------------------------------
// K5: spatial sum of x2 per (bg,o)
// ---------------------------------------------------------------------------
__global__ void __launch_bounds__(256)
k5_m2(const float* __restrict__ x2, float* __restrict__ m2) {
    int bc = blockIdx.x;
    const float4* p = (const float4*)(x2 + (size_t)bc * DHW);
    __shared__ float acc;
    if (threadIdx.x == 0) acc = 0.f;
    __syncthreads();
    float s = 0.f;
    for (int k = threadIdx.x; k < DHW / 4; k += 256) {
        float4 v = p[k];
        s += v.x + v.y + v.z + v.w;
    }
    atomicAdd(&acc, s);
    __syncthreads();
    if (threadIdx.x == 0) m2[bc] = acc;
}

// ---------------------------------------------------------------------------
// K6: x21 = softmax(mean(x2)); x11 = softmax(gn_b)  (spatial-mean of x1 == gn_b)
// ---------------------------------------------------------------------------
__global__ void k6_softmax(const float* __restrict__ m2, const float* __restrict__ gn_b,
                           float* __restrict__ x11, float* __restrict__ x21) {
    int bg = blockIdx.x;
    int t = threadIdx.x;               // 64 threads
    __shared__ float v1[64], v2[64];
    v1[t] = gn_b[t];
    v2[t] = m2[bg * 64 + t] * (1.0f / DHW);
    __syncthreads();
    float m1 = -1e30f, mm2 = -1e30f;
    for (int i = 0; i < 64; ++i) { m1 = fmaxf(m1, v1[i]); mm2 = fmaxf(mm2, v2[i]); }
    float s1 = 0.f, s2 = 0.f;
    for (int i = 0; i < 64; ++i) { s1 += __expf(v1[i] - m1); s2 += __expf(v2[i] - mm2); }
    x11[bg * 64 + t] = __expf(v1[t] - m1) / s1;
    x21[bg * 64 + t] = __expf(v2[t] - mm2) / s2;
}

// ---------------------------------------------------------------------------
// K7: weights = x11.x2 + x21.x1 per spatial point; out = gx * sigmoid(weights).
// x2 aliases out (d_out scratch): all reads happen before the write barrier,
// and blocks touch disjoint rows.
// ---------------------------------------------------------------------------
__global__ void __launch_bounds__(256)
k7_out(const float* __restrict__ x, const float* x2,
       const float* __restrict__ sig, const float* __restrict__ muv,
       const float* __restrict__ rstdv, const float* __restrict__ gn_w,
       const float* __restrict__ gn_b, const float* __restrict__ x11,
       const float* __restrict__ x21, float* out) {
    int bid = blockIdx.x;
    int h = bid & 63;
    int d = (bid >> 6) & 15;
    int bg = bid >> 10;
    int t = threadIdx.x;
    int w = t & 63;
    int chq = t >> 6;                  // 0..3
    __shared__ float xrow[64][64];
    __shared__ float red[4][64];
    __shared__ float sigw[64];
    size_t rowoff = (size_t)bg * 64 * DHW + (size_t)d * HW + (size_t)h * WDIM;
    float part = 0.f;
    for (int cc = 0; cc < 16; ++cc) {
        int ch = chq * 16 + cc;
        size_t off = rowoff + (size_t)ch * DHW + w;
        float xv = x[off];
        float x2v = x2[off];
        xrow[ch][w] = xv;
        int bc = bg * 64 + ch;
        const float* sg = sig + (size_t)bc * SPOOL;
        float g = xv * sg[d] * sg[16 + h] * sg[80 + w];
        float x1v = (g - muv[bc]) * rstdv[bc] * gn_w[ch] + gn_b[ch];
        part += x11[bc] * x2v + x21[bc] * x1v;
    }
    red[chq][w] = part;
    __syncthreads();
    if (t < 64) {
        float s = red[0][t] + red[1][t] + red[2][t] + red[3][t];
        sigw[t] = 1.0f / (1.0f + __expf(-s));
    }
    __syncthreads();
    int ch = t >> 2;
    int w0 = (t & 3) * 16;
    float* ob = out + rowoff + (size_t)ch * DHW;
#pragma unroll
    for (int j = 0; j < 16; ++j) ob[w0 + j] = xrow[ch][w0 + j] * sigw[w0 + j];
}

// ---------------------------------------------------------------------------
extern "C" void kernel_launch(void* const* d_in, const int* in_sizes, int n_in,
                              void* d_out, int out_size, void* d_ws, size_t ws_size,
                              hipStream_t stream) {
    (void)in_sizes; (void)n_in; (void)out_size; (void)ws_size;
    const float* x   = (const float*)d_in[0];
    const float* w1  = (const float*)d_in[1];
    const float* b1  = (const float*)d_in[2];
    const float* w3  = (const float*)d_in[3];
    const float* b3  = (const float*)d_in[4];
    const float* gnw = (const float*)d_in[5];
    const float* gnb = (const float*)d_in[6];
    float* out = (float*)d_out;

    float* ws = (float*)d_ws;
    float* pools = ws;  ws += 1024 * SPOOL;   // 147456
    float* sig   = ws;  ws += 1024 * SPOOL;   // 147456
    float* mu    = ws;  ws += 1024;
    float* rstd  = ws;  ws += 1024;
    float* m2    = ws;  ws += 1024;
    float* x11   = ws;  ws += 1024;
    float* x21   = ws;  ws += 1024;
    unsigned short* afrag = (unsigned short*)ws;  ws += 55296;  // 216 frags * 1KB
    unsigned short* xbf = (unsigned short*)ws;    // 67108864 bf16 = 134 MB
    float* x2 = out;                              // stage conv output in d_out

    hipFuncSetAttribute((const void*)k4_conv,
                        hipFuncAttributeMaxDynamicSharedMemorySize, K4_SMEM_BYTES);

    k0_prep_afrag<<<216, 32, 0, stream>>>(w3, afrag);
    k0b_tobf<<<32768, 256, 0, stream>>>(x, xbf);
    k1_pools<<<1024, 256, 0, stream>>>(x, pools);
    k2_mix<<<16, 256, 0, stream>>>(pools, w1, b1, sig);
    k3_stats<<<1024, 256, 0, stream>>>(x, sig, mu, rstd);
    k4_conv<<<BG * DDIM, 256, K4_SMEM_BYTES, stream>>>(xbf, afrag, b3, x2);
    k5_m2<<<1024, 256, 0, stream>>>(x2, m2);
    k6_softmax<<<BG, 64, 0, stream>>>(m2, gnb, x11, x21);
    k7_out<<<BG * DDIM * HDIM, 256, 0, stream>>>(x, x2, sig, mu, rstd,
                                                 gnw, gnb, x11, x21, out);
}
